// Model_31009663877809
// MI455X (gfx1250) — compile-verified, matrix operand order checked by reading, not run on hardware
//
#include <hip/hip_runtime.h>
#include <hip/hip_bf16.h>
#include <math.h>

typedef __attribute__((ext_vector_type(16))) _Float16 v16h;
typedef __attribute__((ext_vector_type(8)))  float    v8f;
typedef _Float16 f16;
typedef unsigned long long u64;

union HV { v16h v; u64 q[4]; f16 h[16]; };

static __device__ __forceinline__ int imin(int a, int b) { return a < b ? a : b; }

static __device__ __forceinline__ v8f wmma_f16(const HV& a, const HV& b, v8f c) {
  // D = A(16x32 f16) x B(32x16 f16) + C(16x16 f32)
  return __builtin_amdgcn_wmma_f32_16x16x32_f16(false, a.v, false, b.v, (short)0, c,
                                                false, false);
}

// ---------------------------------------------------------------------------
// Problem constants: B=4 C=64 G=8 -> BG=32, cg=8, H=W=260, enc 65x65 x D=128,
// corr 162ch, texture 258x258, tokens = 4*258*258 = 266256.
// ---------------------------------------------------------------------------
#define NTOK 266256L

// ---------------------------------------------------------------------------
// Weight prep: f16 repack of enc_w [128][128], packed conv1 weights
// wk1[16][1472] (n<8: phi_w1, n>=8: vor_w1; K=162*9=1458 zero-padded to 1472),
// fc1/fc2 [64][64] f16.
// ---------------------------------------------------------------------------
__global__ void k_prepw(const float* __restrict__ encw,
                        const float* __restrict__ pw1, const float* __restrict__ vw1,
                        const float* __restrict__ fc1, const float* __restrict__ fc2,
                        f16* __restrict__ wenc, f16* __restrict__ wk1,
                        f16* __restrict__ wfc1, f16* __restrict__ wfc2) {
  int i = blockIdx.x * 256 + threadIdx.x;
  if (i < 16384) wenc[i] = (f16)encw[i];   // [d][c*16+py*4+px] == flat enc_w
  if (i < 23552) {
    int n = i / 1472, kk = i - n * 1472;
    float v = 0.f;
    if (kk < 1458) v = (n < 8 ? pw1 : vw1)[(n & 7) * 1458 + kk]; // [oc][ic*9+ky*3+kx]
    wk1[i] = (f16)v;
  }
  if (i < 4096) { wfc1[i] = (f16)fc1[i]; wfc2[i] = (f16)fc2[i]; }
}

// ---------------------------------------------------------------------------
// Masked f16 inputs: pm = prev*mask, pb = prev*boundary, nm = next*mask.
// Flat layout identical to prev ([4][64][260][260] == [32][8][260][260]).
// ---------------------------------------------------------------------------
__global__ void k_prepmask(const float* __restrict__ prev, const float* __restrict__ next,
                           const float* __restrict__ mask, const float* __restrict__ bnd,
                           f16* __restrict__ pm, f16* __restrict__ pb, f16* __restrict__ nm) {
  long i = (long)blockIdx.x * 256 + threadIdx.x;
  if (i >= 17305600L) return;
  int hw = (int)(i % 67600);
  int b  = (int)(i / (64L * 67600));
  float m  = mask[(long)b * 67600 + hw];
  float bd = bnd [(long)b * 67600 + hw];
  float p = prev[i], n = next[i];
  pm[i] = (f16)(p * m);
  pb[i] = (f16)(p * bd);
  nm[i] = (f16)(n * m);
}

// ---------------------------------------------------------------------------
// Patch encoder as implicit GEMM with WMMA.
// out[bg][p][d] = sum_k src[bg][c][4y+py][4x+px] * w[d][k] + bias[d],
// k = c*16 + py*4 + px, K=128, M=4225 positions, N=128 channels.
// grid = (265 mtiles, 32 bg), block = 256 (8 waves = 8 ntiles).
// ---------------------------------------------------------------------------
__global__ __launch_bounds__(256) void k_encoder(const f16* __restrict__ src,
                                                 const f16* __restrict__ wk,
                                                 const float* __restrict__ bias,
                                                 float* __restrict__ dst) {
  int mtile = blockIdx.x, bg = blockIdx.y;
  int wid = threadIdx.x >> 5, lane = threadIdx.x & 31;
  int l16 = lane & 15;
  bool lo = lane < 16;
  int ntile = wid;
  int p = mtile * 16 + l16;
  int pc = p < 4225 ? p : 4224;
  int y = pc / 65, x = pc - 65 * (pc / 65);
  int py0 = lo ? 0 : 2;
  const f16* abase = src + (size_t)bg * 8 * 67600;
  const f16* wb = wk + (size_t)(ntile * 16 + l16) * 128 + (lo ? 0 : 8);
  v8f acc = {0.f, 0.f, 0.f, 0.f, 0.f, 0.f, 0.f, 0.f};
#pragma unroll
  for (int kb = 0; kb < 128; kb += 32) {
    int c0 = kb >> 4;                       // two input channels per iter
    const f16* ab = abase + ((size_t)c0 * 260 + (4 * y + py0)) * 260 + 4 * x;
    HV A, B;
    A.q[0] = *(const u64*)(ab);             // c0,   py0,   px 0..3
    A.q[1] = *(const u64*)(ab + 260);       // c0,   py0+1
    A.q[2] = *(const u64*)(ab + 67600);     // c0+1, py0
    A.q[3] = *(const u64*)(ab + 67600 + 260);
    const f16* wp = wb + kb;
    B.q[0] = *(const u64*)(wp);
    B.q[1] = *(const u64*)(wp + 4);
    B.q[2] = *(const u64*)(wp + 16);
    B.q[3] = *(const u64*)(wp + 20);
    acc = wmma_f16(A, B, acc);
  }
  int d = ntile * 16 + l16;
  float bv = bias[d];
#pragma unroll
  for (int r = 0; r < 8; ++r) {
    int pp = mtile * 16 + (lo ? r : r + 8);
    if (pp < 4225) dst[((size_t)bg * 4225 + pp) * 128 + d] = acc[r] + bv;
  }
}

// ---------------------------------------------------------------------------
// Correlation: corr[bg][o][y][x], o<81 from (f_pm,f_nm), o>=81 from (f_pb,f_nm);
// mean over 128 channels of f1[y,x,:] * f2[y+dy-4, x+dx-4, :] (zero OOB).
// ---------------------------------------------------------------------------
__global__ void k_corr(const float* __restrict__ fpm, const float* __restrict__ fpb,
                       const float* __restrict__ fnm, float* __restrict__ corr) {
  long i = (long)blockIdx.x * 256 + threadIdx.x;
  if (i >= 21902400L) return;
  int x = (int)(i % 65); long r = i / 65;
  int y = (int)(r % 65); r /= 65;
  int o = (int)(r % 162);
  int bg = (int)(r / 162);
  const float* f1 = (o < 81 ? fpm : fpb) + ((size_t)bg * 4225 + y * 65 + x) * 128;
  int o2 = o < 81 ? o : o - 81;
  int dy = o2 / 9 - 4, dx = o2 - 9 * (o2 / 9) - 4;
  int sy = y + dy, sx = x + dx;
  float s = 0.f;
  if ((unsigned)sy < 65u && (unsigned)sx < 65u) {
    const float4* a = (const float4*)f1;
    const float4* b = (const float4*)(fnm + ((size_t)bg * 4225 + sy * 65 + sx) * 128);
#pragma unroll
    for (int c = 0; c < 32; ++c) {
      float4 u = a[c], v = b[c];
      s += u.x * v.x + u.y * v.y + u.z * v.z + u.w * v.w;
    }
  }
  corr[i] = s * (1.f / 128.f);
}

// ---------------------------------------------------------------------------
// up2x of corr (65 -> 130), stored as f16 for the WMMA conv1 (shared by both
// branches). c = i*(n-1)/(2n-1) bilinear with clamped i1.
// ---------------------------------------------------------------------------
__global__ void k_up1(const float* __restrict__ corr, f16* __restrict__ up1) {
  long i = (long)blockIdx.x * 256 + threadIdx.x;
  if (i >= 87609600L) return;
  int ux = (int)(i % 130); long r = i / 130;
  int uy = (int)(r % 130);
  long plane = r / 130;                     // bg*162 + ic
  float cx = ux * (64.0f / 129.0f);
  float cy = uy * (64.0f / 129.0f);
  int x0 = (int)cx; float wx = cx - x0; int x1 = imin(x0 + 1, 64);
  int y0 = (int)cy; float wy = cy - y0; int y1 = imin(y0 + 1, 64);
  const float* p = corr + plane * 4225;
  float v00 = p[y0 * 65 + x0], v01 = p[y0 * 65 + x1];
  float v10 = p[y1 * 65 + x0], v11 = p[y1 * 65 + x1];
  up1[i] = (f16)((v00 * (1.f - wx) + v01 * wx) * (1.f - wy) +
                 (v10 * (1.f - wx) + v11 * wx) * wy);
}

// ---------------------------------------------------------------------------
// Stage-1 conv of BOTH branches packed: implicit GEMM M=16900 (130x130),
// N=16 (8 phi + 8 vort outputs), K=1472 (162*9 padded). BN + ReLU fused.
// grid = (133, 32 bg), 8 waves/block, wave -> mtile.
// ---------------------------------------------------------------------------
__global__ __launch_bounds__(256) void k_conv1(const f16* __restrict__ up1,
                                               const f16* __restrict__ wk1,
                                               const float* __restrict__ pg, const float* __restrict__ pb,
                                               const float* __restrict__ vg, const float* __restrict__ vb,
                                               float* __restrict__ h1) {
  int wid = threadIdx.x >> 5, lane = threadIdx.x & 31;
  int mtile = blockIdx.x * 8 + wid;
  if (mtile >= 1057) return;                 // whole-wave uniform exit
  int bg = blockIdx.y;
  int l16 = lane & 15;
  bool lo = lane < 16;
  int p = mtile * 16 + l16;
  if (p > 16899) p = 16899;
  int oy = p / 130, ox = p - 130 * (p / 130);
  const f16* src = up1 + (size_t)bg * 162 * 16900;
  const f16* wb = wk1 + (size_t)l16 * 1472 + (lo ? 0 : 8);
  int kadd = lo ? 0 : 8;
  v8f acc = {0.f, 0.f, 0.f, 0.f, 0.f, 0.f, 0.f, 0.f};
  for (int kb = 0; kb < 1472; kb += 32) {
    HV A, B;
#pragma unroll
    for (int h = 0; h < 16; ++h) {
      int k = kb + kadd + ((h >> 3) << 4) + (h & 7);
      f16 val = (f16)0.f;
      if (k < 1458) {
        int ic = k / 9, t = k - ic * 9;
        int ky = t / 3, kx = t - ky * 3;
        int iy = oy + ky - 1, ix = ox + kx - 1;
        if ((unsigned)iy < 130u && (unsigned)ix < 130u)
          val = src[((size_t)ic * 130 + iy) * 130 + ix];
      }
      A.h[h] = val;
    }
    const f16* wp = wb + kb;
    B.q[0] = *(const u64*)(wp);
    B.q[1] = *(const u64*)(wp + 4);
    B.q[2] = *(const u64*)(wp + 16);
    B.q[3] = *(const u64*)(wp + 20);
    acc = wmma_f16(A, B, acc);
  }
  int n = l16, br = n >> 3, ch = n & 7;
  float sc = (br ? vg[ch] : pg[ch]) * rsqrtf(1.0f + 1e-5f);
  float bb = br ? vb[ch] : pb[ch];
  float* dstp = h1 + (((size_t)br * 32 + bg) * 8 + ch) * 16900;
#pragma unroll
  for (int r = 0; r < 8; ++r) {
    int pp = mtile * 16 + (lo ? r : r + 8);
    if (pp < 16900) dstp[pp] = fmaxf(acc[r] * sc + bb, 0.f);
  }
}

// ---------------------------------------------------------------------------
// up2x of h1 branch (130 -> 260), f32.
// ---------------------------------------------------------------------------
__global__ void k_up2(const float* __restrict__ src, float* __restrict__ dst) {
  long i = (long)blockIdx.x * 256 + threadIdx.x;
  if (i >= 17305600L) return;
  int ux = (int)(i % 260); long r = i / 260;
  int uy = (int)(r % 260);
  long plane = r / 260;                     // bg*8 + ch
  float cx = ux * (129.0f / 259.0f);
  float cy = uy * (129.0f / 259.0f);
  int x0 = (int)cx; float wx = cx - x0; int x1 = imin(x0 + 1, 129);
  int y0 = (int)cy; float wy = cy - y0; int y1 = imin(y0 + 1, 129);
  const float* p = src + plane * 16900;
  float v00 = p[y0 * 130 + x0], v01 = p[y0 * 130 + x1];
  float v10 = p[y1 * 130 + x0], v11 = p[y1 * 130 + x1];
  dst[i] = (v00 * (1.f - wx) + v01 * wx) * (1.f - wy) +
           (v10 * (1.f - wx) + v11 * wx) * wy;
}

// ---------------------------------------------------------------------------
// Stage-2 conv 3x3 8->8, pad 1, BN + ReLU (per branch, VALU — only 72 MACs).
// ---------------------------------------------------------------------------
__global__ void k_conv2(const float* __restrict__ U, const float* __restrict__ w,
                        const float* __restrict__ g, const float* __restrict__ bb,
                        float* __restrict__ dst) {
  long i = (long)blockIdx.x * 256 + threadIdx.x;
  if (i >= 17305600L) return;
  int ox = (int)(i % 260); long r = i / 260;
  int oy = (int)(r % 260); r /= 260;
  int oc = (int)(r % 8);
  int bg = (int)(r / 8);
  const float* src = U + (long)bg * 8 * 67600;
  float s = 0.f;
  for (int ic = 0; ic < 8; ++ic) {
    const float* pl = src + (long)ic * 67600;
    const float* wp = w + (oc * 8 + ic) * 9;
#pragma unroll
    for (int ky = 0; ky < 3; ++ky) {
      int iy = oy + ky - 1;
      if ((unsigned)iy >= 260u) continue;
#pragma unroll
      for (int kx = 0; kx < 3; ++kx) {
        int ix = ox + kx - 1;
        if ((unsigned)ix < 260u) s += pl[(long)iy * 260 + ix] * wp[ky * 3 + kx];
      }
    }
  }
  dst[i] = fmaxf(s * (g[oc] * rsqrtf(1.0f + 1e-5f)) + bb[oc], 0.f);
}

// ---------------------------------------------------------------------------
// 1x1 conv heads -> helm[bg][2][260][260] (phi scaled, vort * weight * H).
// ---------------------------------------------------------------------------
__global__ void k_helm(const float* __restrict__ h2p, const float* __restrict__ h2v,
                       const float* __restrict__ pow_, const float* __restrict__ pob,
                       const float* __restrict__ vow, const float* __restrict__ vob,
                       const float* __restrict__ pw, const float* __restrict__ vw,
                       float* __restrict__ helm) {
  long i = (long)blockIdx.x * 256 + threadIdx.x;
  if (i >= 2163200L) return;
  int xy = (int)(i % 67600);
  int bg = (int)(i / 67600);
  const float* hp = h2p + (long)bg * 8 * 67600 + xy;
  const float* hv = h2v + (long)bg * 8 * 67600 + xy;
  float a = 0.f, c = 0.f;
#pragma unroll
  for (int ch = 0; ch < 8; ++ch) {
    a += pow_[ch] * hp[(long)ch * 67600];
    c += vow[ch] * hv[(long)ch * 67600];
  }
  a = (a + pob[0]) * pw[0];
  c = (c + vob[0]) * (vw[0] * 260.0f);
  helm[(long)bg * 2 * 67600 + xy] = a;
  helm[(long)bg * 2 * 67600 + 67600 + xy] = c;
}

// ---------------------------------------------------------------------------
// Velocity: phi central differences + vortex-kernel stencil, combined.
//   u_phi = 0.5*(P[y+1][x+2]-P[y+1][x]);  v_phi = 0.5*(P[y+2][x+1]-P[y][x+1])
//   u_vort += 0.5*K0[i][j]*(s[y+i-2][x+j-2]+s[y+i-2][x+j-3]);  (s = S[:-1,:-1])
//   v_vort += 0.5*K1[i][j]*(s[y+i-2][x+j-2]+s[y+i-3][x+j-2])
// (kernel weights constant-folded by unrolling.)
// ---------------------------------------------------------------------------
__global__ void k_vel(const float* __restrict__ helm, float* __restrict__ vel) {
  long i = (long)blockIdx.x * 256 + threadIdx.x;
  if (i >= 2130048L) return;
  int x = (int)(i % 258); long r = i / 258;
  int y = (int)(r % 258);
  int bg = (int)(r / 258);
  const float* P = helm + (long)bg * 2 * 67600;
  const float* S = P + 67600;
  float u = 0.5f * (P[(y + 1) * 260 + (x + 2)] - P[(y + 1) * 260 + x]);
  float v = 0.5f * (P[(y + 2) * 260 + (x + 1)] - P[y * 260 + (x + 1)]);
#pragma unroll
  for (int i6 = 0; i6 < 6; ++i6) {
#pragma unroll
    for (int j6 = 0; j6 < 6; ++j6) {
      float dy = i6 - 2.5f, dx = j6 - 2.5f;
      float inv = 1.0f / (6.28318530718f * (dy * dy + dx * dx));
      float k0 = -dy * inv, k1 = dx * inv;
      int ay = y + i6 - 2, ax = x + j6 - 2;
      float a  = ((unsigned)ay < 259u && (unsigned)ax < 259u) ? S[ay * 260 + ax] : 0.f;
      float al = ((unsigned)ay < 259u && (unsigned)(ax - 1) < 259u) ? S[ay * 260 + ax - 1] : 0.f;
      float au = ((unsigned)(ay - 1) < 259u && (unsigned)ax < 259u) ? S[(ay - 1) * 260 + ax] : 0.f;
      u += 0.5f * k0 * (a + al);
      v += 0.5f * k1 * (a + au);
    }
  }
  vel[(long)bg * 2 * 66564 + y * 258 + x] = u;
  vel[(long)bg * 2 * 66564 + 66564 + y * 258 + x] = v;
}

// ---------------------------------------------------------------------------
// Bilinear back-warp. mode 0: dst[bg][c][y][x] = sample(src).
// mode 1 (final BFECC combine, exploiting linearity of bilinear sampling):
//   pred_t[token][g*8+c] = 1.5*f1 - 0.5*sample(src)   (channel-last for MLP)
// ---------------------------------------------------------------------------
__global__ void k_warp(const float* __restrict__ src, const float* __restrict__ vel,
                       float sign, int mode, const float* __restrict__ f1,
                       float* __restrict__ dst) {
  long i = (long)blockIdx.x * 256 + threadIdx.x;
  if (i >= 2130048L) return;
  int x = (int)(i % 258); long r = i / 258;
  int y = (int)(r % 258);
  int bg = (int)(r / 258);
  long vbase = (long)bg * 2 * 66564 + y * 258 + x;
  float px = (float)x - sign * vel[vbase];
  float py = (float)y - sign * vel[vbase + 66564];
  px = fminf(fmaxf(px, 0.f), 257.f);
  py = fminf(fmaxf(py, 0.f), 257.f);
  float fx = floorf(px), fy = floorf(py);
  float wx = px - fx, wy = py - fy;
  int x0 = (int)fx, y0 = (int)fy;
  int x1 = imin(x0 + 1, 257), y1 = imin(y0 + 1, 257);
  int b = bg >> 3, g_ = bg & 7;
  for (int c = 0; c < 8; ++c) {
    const float* f = src + ((long)bg * 8 + c) * 66564;
    float v00 = f[y0 * 258 + x0], v01 = f[y0 * 258 + x1];
    float v10 = f[y1 * 258 + x0], v11 = f[y1 * 258 + x1];
    float sv = (v00 * (1.f - wx) + v01 * wx) * (1.f - wy) +
               (v10 * (1.f - wx) + v11 * wx) * wy;
    if (mode == 0) {
      dst[((long)bg * 8 + c) * 66564 + y * 258 + x] = sv;
    } else {
      long tok = (long)b * 66564 + y * 258 + x;
      dst[tok * 64 + (g_ * 8 + c)] =
          1.5f * f1[((long)bg * 8 + c) * 66564 + y * 258 + x] - 0.5f * sv;
    }
  }
}

// ---------------------------------------------------------------------------
// LayerNorm + MLP (64 -> gelu -> 64) + residual, WMMA from LDS.
// Block: 128 tokens, 8 waves; 8 mtiles x 4 ntiles = 32 wave-tiles, 4 per wave.
// ---------------------------------------------------------------------------
__global__ __launch_bounds__(256) void k_mlp(const float* __restrict__ predt,
                                             const float* __restrict__ lng,
                                             const float* __restrict__ lnbias,
                                             const f16* __restrict__ w1, const float* __restrict__ b1,
                                             const f16* __restrict__ w2, const float* __restrict__ b2,
                                             float* __restrict__ out) {
  __shared__ f16 lnS[128 * 64];
  __shared__ f16 hS[128 * 64];
  int tid = threadIdx.x;
  long t0 = (long)blockIdx.x * 128;
  if (tid < 128) {
    long t = t0 + tid;
    if (t > NTOK - 1) t = NTOK - 1;
    const float* xp = predt + t * 64;
    float s = 0.f, s2 = 0.f;
#pragma unroll
    for (int c = 0; c < 64; ++c) { float v = xp[c]; s += v; s2 += v * v; }
    float mu = s * (1.f / 64.f);
    float var = s2 * (1.f / 64.f) - mu * mu;
    float inv = rsqrtf(var + 1e-5f);
#pragma unroll
    for (int c = 0; c < 64; ++c)
      lnS[tid * 64 + c] = (f16)((xp[c] - mu) * inv * lng[c] + lnbias[c]);
  }
  __syncthreads();
  int wid = tid >> 5, lane = tid & 31, l16 = lane & 15;
  bool lo = lane < 16;
  int kofs = lo ? 0 : 8;
  // GEMM1 + exact GELU
#pragma unroll
  for (int it = 0; it < 4; ++it) {
    int tile = wid * 4 + it;
    int mt = tile >> 2, nt = tile & 3;
    const f16* arow = &lnS[(mt * 16 + l16) * 64 + kofs];
    const f16* brow = w1 + (long)(nt * 16 + l16) * 64 + kofs;
    v8f acc = {0.f, 0.f, 0.f, 0.f, 0.f, 0.f, 0.f, 0.f};
#pragma unroll
    for (int kb = 0; kb < 64; kb += 32) {
      HV A, B;
      A.q[0] = *(const u64*)(arow + kb);      A.q[1] = *(const u64*)(arow + kb + 4);
      A.q[2] = *(const u64*)(arow + kb + 16); A.q[3] = *(const u64*)(arow + kb + 20);
      B.q[0] = *(const u64*)(brow + kb);      B.q[1] = *(const u64*)(brow + kb + 4);
      B.q[2] = *(const u64*)(brow + kb + 16); B.q[3] = *(const u64*)(brow + kb + 20);
      acc = wmma_f16(A, B, acc);
    }
    int n = nt * 16 + l16;
    float bv = b1[n];
#pragma unroll
    for (int rr = 0; rr < 8; ++rr) {
      int row = mt * 16 + (lo ? rr : rr + 8);
      float v = acc[rr] + bv;
      v = 0.5f * v * (1.0f + erff(v * 0.70710678118654752f));
      hS[row * 64 + n] = (f16)v;
    }
  }
  __syncthreads();
  // GEMM2 + residual, write NCHW output
#pragma unroll
  for (int it = 0; it < 4; ++it) {
    int tile = wid * 4 + it;
    int mt = tile >> 2, nt = tile & 3;
    const f16* arow = &hS[(mt * 16 + l16) * 64 + kofs];
    const f16* brow = w2 + (long)(nt * 16 + l16) * 64 + kofs;
    v8f acc = {0.f, 0.f, 0.f, 0.f, 0.f, 0.f, 0.f, 0.f};
#pragma unroll
    for (int kb = 0; kb < 64; kb += 32) {
      HV A, B;
      A.q[0] = *(const u64*)(arow + kb);      A.q[1] = *(const u64*)(arow + kb + 4);
      A.q[2] = *(const u64*)(arow + kb + 16); A.q[3] = *(const u64*)(arow + kb + 20);
      B.q[0] = *(const u64*)(brow + kb);      B.q[1] = *(const u64*)(brow + kb + 4);
      B.q[2] = *(const u64*)(brow + kb + 16); B.q[3] = *(const u64*)(brow + kb + 20);
      acc = wmma_f16(A, B, acc);
    }
    int n = nt * 16 + l16;
    float bv = b2[n];
#pragma unroll
    for (int rr = 0; rr < 8; ++rr) {
      int row = mt * 16 + (lo ? rr : rr + 8);
      long t = t0 + row;
      if (t < NTOK) {
        long b = t / 66564, rem = t - b * 66564;
        out[(b * 64 + n) * 66564 + rem] = predt[t * 64 + n] + (acc[rr] + bv);
      }
    }
  }
}

// ---------------------------------------------------------------------------
// Host launcher. Workspace layout (bytes, lifetime-based reuse, ~503 MB):
//   [0 .. 207.7M)   : feats f_pm/f_pb/f_nm  ->  up1(f16)  ->  f1/fb/pred_t
//   [207.7 .. 295.3M): corr  ->  up2 scratch
//   [295.3 .. 329.9M): h1[2]
//   [329.9 .. 468.3M): pm16/pb16/nm16 (phase 1)  ->  h2[2]
//   [468.3 .. 485.6M): helm       [485.6 .. 502.7M): vel
//   [502.7M ..)      : f16 weights (~96 KB)
// ---------------------------------------------------------------------------
extern "C" void kernel_launch(void* const* d_in, const int* in_sizes, int n_in,
                              void* d_out, int out_size, void* d_ws, size_t ws_size,
                              hipStream_t stream) {
  (void)in_sizes; (void)n_in; (void)out_size; (void)ws_size;
  const float* prev     = (const float*)d_in[0];
  const float* next     = (const float*)d_in[1];
  const float* texture  = (const float*)d_in[2];
  const float* mask     = (const float*)d_in[3];
  const float* boundary = (const float*)d_in[4];
  const float* enc_w    = (const float*)d_in[5];
  const float* enc_b    = (const float*)d_in[6];
  const float* phi_w1   = (const float*)d_in[7];
  const float* phi_g1   = (const float*)d_in[8];
  const float* phi_b1   = (const float*)d_in[9];
  const float* phi_w2   = (const float*)d_in[10];
  const float* phi_g2   = (const float*)d_in[11];
  const float* phi_b2   = (const float*)d_in[12];
  const float* phi_ow   = (const float*)d_in[13];
  const float* phi_ob   = (const float*)d_in[14];
  const float* vor_w1   = (const float*)d_in[15];
  const float* vor_g1   = (const float*)d_in[16];
  const float* vor_b1   = (const float*)d_in[17];
  const float* vor_w2   = (const float*)d_in[18];
  const float* vor_g2   = (const float*)d_in[19];
  const float* vor_b2   = (const float*)d_in[20];
  const float* vor_ow   = (const float*)d_in[21];
  const float* vor_ob   = (const float*)d_in[22];
  const float* phi_wt   = (const float*)d_in[23];
  const float* vor_wt   = (const float*)d_in[24];
  const float* ln_g     = (const float*)d_in[25];
  const float* ln_b     = (const float*)d_in[26];
  const float* fc1_w    = (const float*)d_in[27];
  const float* fc1_b    = (const float*)d_in[28];
  const float* fc2_w    = (const float*)d_in[29];
  const float* fc2_b    = (const float*)d_in[30];

  char* W = (char*)d_ws;
  float* fpm  = (float*)(W + 0);
  float* fpb  = (float*)(W + 69222400LL);
  float* fnm  = (float*)(W + 138444800LL);
  f16*   up1  = (f16*)  (W + 0);            // reuses feats region (after corr)
  float* corr = (float*)(W + 207667200LL);
  float* h1   = (float*)(W + 295276800LL);  // [2][32][8][130][130]
  f16*   pm16 = (f16*)  (W + 329888000LL);
  f16*   pb16 = (f16*)  (W + 364499200LL);
  f16*   nm16 = (f16*)  (W + 399110400LL);
  float* Ubuf = (float*)(W + 207667200LL);  // reuses corr region (after conv1)
  float* h2   = (float*)(W + 329888000LL);  // [2][32][8][260][260], reuses pm16..
  float* helm = (float*)(W + 468332800LL);
  float* vel  = (float*)(W + 485638400LL);
  float* f1b  = (float*)(W + 0);            // reuses up1 region (after conv1)
  float* fbb  = (float*)(W + 68161536LL);
  float* predt= (float*)(W + 136323072LL);  // [266256][64] channel-last
  f16*   wenc = (f16*)  (W + 502678784LL);
  f16*   wk1  = (f16*)  (W + 502711552LL);
  f16*   wfc1 = (f16*)  (W + 502758656LL);
  f16*   wfc2 = (f16*)  (W + 502766848LL);

  float* out = (float*)d_out;
  const long H1_BR = 32L * 8 * 16900;       // h1 branch stride (floats)
  const long H2_BR = 32L * 8 * 67600;       // h2 branch stride (floats)

  k_prepw<<<92, 256, 0, stream>>>(enc_w, phi_w1, vor_w1, fc1_w, fc2_w,
                                  wenc, wk1, wfc1, wfc2);
  k_prepmask<<<67600, 256, 0, stream>>>(prev, next, mask, boundary, pm16, pb16, nm16);

  dim3 ge(265, 32);
  k_encoder<<<ge, 256, 0, stream>>>(pm16, wenc, enc_b, fpm);
  k_encoder<<<ge, 256, 0, stream>>>(pb16, wenc, enc_b, fpb);
  k_encoder<<<ge, 256, 0, stream>>>(nm16, wenc, enc_b, fnm);

  k_corr<<<85557, 256, 0, stream>>>(fpm, fpb, fnm, corr);
  k_up1<<<342225, 256, 0, stream>>>(corr, up1);

  dim3 gc1(133, 32);
  k_conv1<<<gc1, 256, 0, stream>>>(up1, wk1, phi_g1, phi_b1, vor_g1, vor_b1, h1);

  // branch 0 (phi), branch 1 (vort): stage-2 upsample + conv, Ubuf reused
  k_up2<<<67600, 256, 0, stream>>>(h1, Ubuf);
  k_conv2<<<67600, 256, 0, stream>>>(Ubuf, phi_w2, phi_g2, phi_b2, h2);
  k_up2<<<67600, 256, 0, stream>>>(h1 + H1_BR, Ubuf);
  k_conv2<<<67600, 256, 0, stream>>>(Ubuf, vor_w2, vor_g2, vor_b2, h2 + H2_BR);

  k_helm<<<8450, 256, 0, stream>>>(h2, h2 + H2_BR, phi_ow, phi_ob, vor_ow, vor_ob,
                                   phi_wt, vor_wt, helm);
  k_vel<<<8321, 256, 0, stream>>>(helm, vel);

  // BFECC: f1 = warp(tex,+1); fb = warp(f1,-1); pred = 1.5*f1 - 0.5*warp(fb,+1)
  k_warp<<<8321, 256, 0, stream>>>(texture, vel,  1.0f, 0, f1b, f1b);
  k_warp<<<8321, 256, 0, stream>>>(f1b,     vel, -1.0f, 0, f1b, fbb);
  k_warp<<<8321, 256, 0, stream>>>(fbb,     vel,  1.0f, 1, f1b, predt);

  k_mlp<<<2081, 256, 0, stream>>>(predt, ln_g, ln_b, wfc1, fc1_b, wfc2, fc2_b, out);
}
